// NeuralSDE_11596411699493
// MI455X (gfx1250) — matrix operand order repaired and checked by reading, not compiled, not run on hardware
//
#include <hip/hip_runtime.h>
#include <hip/hip_bf16.h>
#include <math.h>

typedef __attribute__((ext_vector_type(16))) _Float16 v16h;
typedef __attribute__((ext_vector_type(8)))  _Float16 v8h;
typedef __attribute__((ext_vector_type(8)))  float    v8f;

#define STATE_DIM 4
#define NOISE_DIM 4
#define HIDDEN    64
#define T_LEN     2049
#define N_PATH    1024
#define NSTEP     2048          // pairs per path
#define TILES_PER_PATH 128      // 2048 / 16
#define WAVES_PER_BLOCK 8
#define BLOCK_THREADS 256

// ---- LDS layout (single char arena, < 64KB) --------------------------------
#define OFF_FRAG 0
#define SZ_FRAG  (36*512*2)              // 36864
#define OFF_W0   (OFF_FRAG + SZ_FRAG)    // 36864
#define SZ_W0    (2*256*4)               // 2048
#define OFF_BIAS (OFF_W0 + SZ_W0)        // 38912
#define SZ_BIAS  (2*208*4)               // 1664
#define OFF_ZB   (OFF_BIAS + SZ_BIAS)    // 40576
#define SZ_ZB    (8*64*4)                // 2048
#define OFF_RED  (OFF_ZB + SZ_ZB)        // 42624
#define SZ_RED   (256*4)                 // 1024
#define OFF_H    (OFF_RED + SZ_RED)      // 43648
#define SZ_H     (8*1024*2)              // 16384
#define SMEM_TOTAL (OFF_H + SZ_H)        // 60032 bytes

__device__ __forceinline__ void waitds() {
    asm volatile("s_wait_dscnt 0" ::: "memory");
}

// Branch-free tanh. gfx1250 has V_TANH_F32 (single TRANS op, co-executes with
// the WMMA pipe); otherwise exp2/rcp form: 1 - 2/(e^{2x}+1), robust at +-inf.
__device__ __forceinline__ float fast_tanh(float x) {
#if __has_builtin(__builtin_amdgcn_tanhf)
    return __builtin_amdgcn_tanhf(x);
#else
    float e = __builtin_amdgcn_exp2f(x * 2.8853900817779268f); // 2*log2(e)
    return 1.0f - 2.0f * __builtin_amdgcn_rcpf(e + 1.0f);
#endif
}

__device__ __forceinline__ v8f splat8(float v) {
    v8f c;
#pragma unroll
    for (int i = 0; i < 8; ++i) c[i] = v;
    return c;
}

// A fragment (16x32 f16): lane l holds row M = l%16,
// K = k*32 + (l/16)*8 + (e<8 ? e : e+8)  -> two contiguous 16B chunks.
__device__ __forceinline__ v16h load_afrag(const _Float16* H, int lane, int k) {
    int row = lane & 15;
    int K0  = k * 32 + ((lane >> 4) << 3);
    const _Float16* p = H + row * 64 + K0;
    v8h lo = *(const v8h*)p;
    v8h hi = *(const v8h*)(p + 16);
    v16h a;
#pragma unroll
    for (int i = 0; i < 8; ++i) { a[i] = lo[i]; a[i + 8] = hi[i]; }
    return a;
}

__device__ __forceinline__ v16h load_bfrag(const _Float16* fragBase, int lane) {
    return *(const v16h*)(fragBase + lane * 16);
}

// One 64->64 hidden layer on a 16-sample tile, in place in H (f16 [16][64]).
__device__ __forceinline__ void hidden_layer(_Float16* H, const _Float16* frags,
                                             const float* bias, int lane) {
    int col16 = lane & 15;
    int rb    = (lane >> 4) * 8;
    v16h a0 = load_afrag(H, lane, 0);
    v16h a1 = load_afrag(H, lane, 1);
#pragma unroll
    for (int j = 0; j < 4; ++j) {
        v8f c = splat8(bias[j * 16 + col16]);
        v16h b0 = load_bfrag(frags + (j * 2 + 0) * 512, lane);
        c = __builtin_amdgcn_wmma_f32_16x16x32_f16(false, a0, false, b0,
                                                   (short)0, c, false, false);
        v16h b1 = load_bfrag(frags + (j * 2 + 1) * 512, lane);
        c = __builtin_amdgcn_wmma_f32_16x16x32_f16(false, a1, false, b1,
                                                   (short)0, c, false, false);
#pragma unroll
        for (int r = 0; r < 8; ++r)
            H[(r + rb) * 64 + j * 16 + col16] = (_Float16)fast_tanh(c[r]);
    }
    waitds();
}

__global__ void __launch_bounds__(BLOCK_THREADS)
sde_nll_kernel(const float* __restrict__ x, const float* __restrict__ hptr,
               const float* __restrict__ dW0, const float* __restrict__ db0,
               const float* __restrict__ dW1, const float* __restrict__ db1,
               const float* __restrict__ dW2, const float* __restrict__ db2,
               const float* __restrict__ dWo, const float* __restrict__ dbo,
               const float* __restrict__ sW0, const float* __restrict__ sb0,
               const float* __restrict__ sW1, const float* __restrict__ sb1,
               const float* __restrict__ sW2, const float* __restrict__ sb2,
               const float* __restrict__ sWo, const float* __restrict__ sbo,
               float* __restrict__ partial) {
    __shared__ __align__(32) unsigned char smem[SMEM_TOTAL];
    _Float16* fragS = (_Float16*)(smem + OFF_FRAG);
    float*    W0s   = (float*)(smem + OFF_W0);
    float*    biasS = (float*)(smem + OFF_BIAS);
    float*    zbufS = (float*)(smem + OFF_ZB);
    float*    redS  = (float*)(smem + OFF_RED);
    _Float16* HbufS = (_Float16*)(smem + OFF_H);

    const int tid = threadIdx.x;

    // ---- build pre-swizzled B fragments in LDS -----------------------------
    // frag element (lane, e): K_local = (lane/16)*8 + (e<8 ? e : e+8), col = 16*j + lane%16
    for (int idx = tid; idx < 36 * 512; idx += BLOCK_THREADS) {
        int f    = idx >> 9;
        int rem  = idx & 511;
        int lane = rem >> 4;
        int e    = rem & 15;
        int m    = f / 18;
        int fid  = f % 18;
        int col16 = lane & 15;
        int Kl    = ((lane >> 4) << 3) + (e & 7) + ((e >> 3) << 4);      // 0..31
        float val = 0.0f;
        if (fid < 8) {                            // layer 1 (64x64)
            int j = fid >> 1, k = fid & 1;
            const float* W = m ? sW1 : dW1;
            val = W[(k * 32 + Kl) * 64 + j * 16 + col16];
        } else if (fid < 16) {                    // layer 2 (64x64)
            int j = (fid - 8) >> 1, k = (fid - 8) & 1;
            const float* W = m ? sW2 : dW2;
            val = W[(k * 32 + Kl) * 64 + j * 16 + col16];
        } else {                                  // output layer
            int k = fid - 16;
            if (m) {
                val = sWo[(k * 32 + Kl) * 16 + col16];        // 64x16
            } else if (col16 < 4) {
                val = dWo[(k * 32 + Kl) * 4 + col16];         // 64x4 padded
            }
        }
        fragS[idx] = (_Float16)val;
    }
    for (int idx = tid; idx < 512; idx += BLOCK_THREADS) {
        int m = idx / 256, o = idx % 256;
        W0s[idx] = (m ? sW0 : dW0)[o];
    }
    for (int idx = tid; idx < 416; idx += BLOCK_THREADS) {
        int m = idx / 208, o = idx % 208;
        float v = 0.0f;
        if (o < 64)       v = (m ? sb0 : db0)[o];
        else if (o < 128) v = (m ? sb1 : db1)[o - 64];
        else if (o < 192) v = (m ? sb2 : db2)[o - 128];
        else { int c = o - 192; if (m) v = sbo[c]; else if (c < 4) v = dbo[c]; }
        biasS[idx] = v;
    }
    __syncthreads();

    const int wave  = tid >> 5;
    const int lane  = tid & 31;
    const int col16 = lane & 15;
    const int rb    = (lane >> 4) * 8;
    const int path  = blockIdx.x;
    const float hval = *hptr;

    _Float16* Hw = HbufS + wave * 1024;
    float*    sb = (float*)Hw;           // sigma staging aliases H (used after H)
    float*    zb = zbufS + wave * 64;

    float acc = 0.0f;

    for (int tile = wave; tile < TILES_PER_PATH; tile += WAVES_PER_BLOCK) {
        int t0 = tile * 16;
        // x1 row for this lane (lanes 16..31 mirror rows 0..15; used in L0)
        const float* xp = x + (((size_t)path * T_LEN) + t0 + col16) * 4;
        float4 x1v = *(const float4*)xp;
        float xr[4] = {x1v.x, x1v.y, x1v.z, x1v.w};

#pragma unroll
        for (int m = 0; m < 2; ++m) {
            const _Float16* fb = fragS + m * 18 * 512;
            const float*    w0 = W0s + m * 256;
            const float*    bb = biasS + m * 208;

            // ---- layer 0 (4 -> 64), VALU; lane covers 32 features of 1 row
            int fbase = (lane >> 4) * 32;
#pragma unroll
            for (int ff = 0; ff < 32; ++ff) {
                int f = fbase + ff;
                float a = bb[f] + xr[0] * w0[f] + xr[1] * w0[64 + f]
                                + xr[2] * w0[128 + f] + xr[3] * w0[192 + f];
                Hw[col16 * 64 + f] = (_Float16)fast_tanh(a);
            }
            waitds();

            // ---- hidden layers via WMMA
            hidden_layer(Hw, fb,            bb + 64,  lane);
            hidden_layer(Hw, fb + 8 * 512,  bb + 128, lane);

            // ---- output layer (64 -> 16, drift padded)
            v16h a0 = load_afrag(Hw, lane, 0);
            v16h a1 = load_afrag(Hw, lane, 1);
            v8f  c  = splat8(bb[192 + col16]);
            v16h b0 = load_bfrag(fb + 16 * 512, lane);
            c = __builtin_amdgcn_wmma_f32_16x16x32_f16(false, a0, false, b0,
                                                       (short)0, c, false, false);
            v16h b1 = load_bfrag(fb + 17 * 512, lane);
            c = __builtin_amdgcn_wmma_f32_16x16x32_f16(false, a1, false, b1,
                                                       (short)0, c, false, false);
            if (m == 0) {
                if (col16 < 4) {
#pragma unroll
                    for (int r = 0; r < 8; ++r)
                        zb[(r + rb) * 4 + col16] = c[r];
                }
            } else {
#pragma unroll
                for (int r = 0; r < 8; ++r)
                    sb[(r + rb) * 16 + col16] = c[r];
            }
            waitds();
        }

        // ---- per-sample epilogue: lanes 0..15, one sample each -------------
        float v = 0.0f;
        if (lane < 16) {
            const float* x2p = x + (((size_t)path * T_LEN) + t0 + lane + 1) * 4;
            float4 x2v = *(const float4*)x2p;
            float zv[4];
            zv[0] = x2v.x - xr[0] - hval * zb[lane * 4 + 0];
            zv[1] = x2v.y - xr[1] - hval * zb[lane * 4 + 1];
            zv[2] = x2v.z - xr[2] - hval * zb[lane * 4 + 2];
            zv[3] = x2v.w - xr[3] - hval * zb[lane * 4 + 3];

            float sg[16];
#pragma unroll
            for (int i = 0; i < 16; ++i) sg[i] = sb[lane * 16 + i];

            // Sigma = h * sigma @ sigma^T  (4x4 SPD)
            float S[4][4];
#pragma unroll
            for (int i = 0; i < 4; ++i)
#pragma unroll
                for (int j = 0; j < 4; ++j)
                    S[i][j] = hval * (sg[4 * i + 0] * sg[4 * j + 0] +
                                      sg[4 * i + 1] * sg[4 * j + 1] +
                                      sg[4 * i + 2] * sg[4 * j + 2] +
                                      sg[4 * i + 3] * sg[4 * j + 3]);

            // Cholesky: S = L L^T
            float L[4][4];
#pragma unroll
            for (int i = 0; i < 4; ++i) {
#pragma unroll
                for (int j = 0; j < 4; ++j) {
                    if (j > i) continue;
                    float s2 = S[i][j];
#pragma unroll
                    for (int k = 0; k < 4; ++k)
                        if (k < j) s2 -= L[i][k] * L[j][k];
                    if (i == j) L[i][i] = sqrtf(fmaxf(s2, 1e-30f));
                    else        L[i][j] = s2 / L[j][j];
                }
            }
            float logdet = 2.0f * logf(L[0][0] * L[1][1] * L[2][2] * L[3][3]);

            // quad = || L^{-1} z ||^2
            float y[4];
#pragma unroll
            for (int i = 0; i < 4; ++i) {
                float t = zv[i];
#pragma unroll
                for (int k = 0; k < 4; ++k)
                    if (k < i) t -= L[i][k] * y[k];
                y[i] = t / L[i][i];
            }
            float quad = y[0] * y[0] + y[1] * y[1] + y[2] * y[2] + y[3] * y[3];
            v = 0.5f * (quad + logdet);
        }
        acc += v;
    }

    // ---- block reduction ---------------------------------------------------
    redS[tid] = acc;
    __syncthreads();
#pragma unroll
    for (int off = 128; off > 0; off >>= 1) {
        if (tid < off) redS[tid] += redS[tid + off];
        __syncthreads();
    }
    if (tid == 0) partial[blockIdx.x] = redS[0];
}

__global__ void __launch_bounds__(256)
reduce_final(const float* __restrict__ partial, float* __restrict__ out, int n) {
    __shared__ float r[256];
    float s = 0.0f;
    for (int i = threadIdx.x; i < n; i += 256) s += partial[i];
    r[threadIdx.x] = s;
    __syncthreads();
#pragma unroll
    for (int off = 128; off > 0; off >>= 1) {
        if (threadIdx.x < off) r[threadIdx.x] += r[threadIdx.x + off];
        __syncthreads();
    }
    if (threadIdx.x == 0) out[0] = r[0];
}

extern "C" void kernel_launch(void* const* d_in, const int* in_sizes, int n_in,
                              void* d_out, int out_size, void* d_ws, size_t ws_size,
                              hipStream_t stream) {
    (void)in_sizes; (void)n_in; (void)out_size; (void)ws_size;
    const float* x    = (const float*)d_in[0];
    const float* h    = (const float*)d_in[1];
    const float* dW0p = (const float*)d_in[2];
    const float* db0p = (const float*)d_in[3];
    const float* dW1p = (const float*)d_in[4];
    const float* db1p = (const float*)d_in[5];
    const float* dW2p = (const float*)d_in[6];
    const float* db2p = (const float*)d_in[7];
    const float* dWop = (const float*)d_in[8];
    const float* dbop = (const float*)d_in[9];
    const float* sW0p = (const float*)d_in[10];
    const float* sb0p = (const float*)d_in[11];
    const float* sW1p = (const float*)d_in[12];
    const float* sb1p = (const float*)d_in[13];
    const float* sW2p = (const float*)d_in[14];
    const float* sb2p = (const float*)d_in[15];
    const float* sWop = (const float*)d_in[16];
    const float* sbop = (const float*)d_in[17];

    float* partial = (float*)d_ws;     // N_PATH floats of scratch
    sde_nll_kernel<<<N_PATH, BLOCK_THREADS, 0, stream>>>(
        x, h, dW0p, db0p, dW1p, db1p, dW2p, db2p, dWop, dbop,
        sW0p, sb0p, sW1p, sb1p, sW2p, sb2p, sWop, sbop, partial);
    reduce_final<<<1, 256, 0, stream>>>(partial, (float*)d_out, N_PATH);
}
